// OneHotRNN_60155311947869
// MI455X (gfx1250) — matrix operand order repaired and checked by reading, not compiled, not run on hardware
//
#include <hip/hip_runtime.h>

#define B_  512
#define T_  512
#define V_  96
#define H_  128
#define BT  16    // batch rows per workgroup (one 16-row M tile)
#define HP  136   // padded LDS row stride in halves (272 B -> bank-conflict-free)

typedef __attribute__((ext_vector_type(16))) _Float16 v16h;
typedef __attribute__((ext_vector_type(8)))  float    v8f;

// ---------------------------------------------------------------------------
// Kernel 1: one-hot -> u16 index. One wave32 per (b,t) row of 96 floats.
// Lane l reads x[l], x[l+32], x[l+64] (fully coalesced), shuffle-max reduce.
// 100 MB read @ 23.3 TB/s ~ 4.4 us : this is the whole HBM cost of the problem.
// ---------------------------------------------------------------------------
__global__ __launch_bounds__(256) void onehot_argmax_kernel(
    const float* __restrict__ x, unsigned short* __restrict__ idx)
{
    const int gtid = blockIdx.x * 256 + threadIdx.x;
    const int row  = gtid >> 5;       // exact: grid = B*T/8 blocks
    const int lane = gtid & 31;
    const float* r = x + (size_t)row * V_;
    int cand = 0;                     // idx==0 case: no lane fires, cand stays 0
#pragma unroll
    for (int c = 0; c < 3; ++c) {
        int v = lane + 32 * c;
        if (r[v] > 0.5f) cand = v;
    }
#pragma unroll
    for (int off = 16; off > 0; off >>= 1)
        cand = max(cand, __shfl_xor(cand, off, 32));
    if (lane == 0) idx[row] = (unsigned short)cand;
}

__device__ __forceinline__ float tanh_fast(float x) {
    float e = __expf(2.0f * x);                       // v_exp_f32
    return 1.0f - 2.0f * __builtin_amdgcn_rcpf(e + 1.0f);
}

// ---------------------------------------------------------------------------
// Kernel 2: sequential RNN scan, WMMA f32<-f16, plus fused final FC.
// 32 WGs x 8 waves. Wave w owns output columns [16w,16w+16) of h.
// W_hh^T B-fragments live in VGPRs for the whole scan; h ping-pongs in LDS.
// ---------------------------------------------------------------------------
__global__ __launch_bounds__(256) void rnn_scan_kernel(
    const unsigned short* __restrict__ idx,
    const float* __restrict__ W_ih, const float* __restrict__ b_ih,
    const float* __restrict__ W_hh, const float* __restrict__ b_hh,
    const float* __restrict__ W_fc, const float* __restrict__ b_fc,
    float* __restrict__ out)
{
    __shared__ _Float16       WihT[V_][H_];        // 24 KB: W_ih^T as f16 (char-major)
    __shared__ float          bsum[H_];            // b_ih + b_hh
    __shared__ __align__(16) unsigned short idxl[T_][BT]; // 16 KB, transposed:
                                                   // step t's 8 indices = 1 b128 load
    __shared__ __align__(16) _Float16 hbuf[2][BT][HP];    // 8.5 KB ping-pong h

    const int tid  = threadIdx.x;
    const int lane = tid & 31;
    const int wave = tid >> 5;
    const int rb   = blockIdx.x * BT;

    // ---- stage constants into LDS -----------------------------------------
    for (int i = tid; i < V_ * H_; i += 256) {
        int v = i >> 7, h = i & (H_ - 1);
        WihT[v][h] = (_Float16)W_ih[h * V_ + v];
    }
    if (tid < H_) bsum[tid] = b_ih[tid] + b_hh[tid];
    for (int i = tid; i < BT * T_; i += 256) {     // coalesced global along t
        int m = i >> 9, t = i & (T_ - 1);
        idxl[t][m] = idx[(size_t)(rb + m) * T_ + t];
    }
    for (int i = tid; i < BT * HP; i += 256)       // h0 = 0 (buffer 0)
        (&hbuf[0][0][0])[i] = (_Float16)0.0f;

    // ---- per-lane WMMA geometry (ISA 16x16x32 f16 layouts) ----------------
    const int ncol = wave * 16 + (lane & 15);      // B/C/D column N
    const int hiK  = (lane & 16) ? 16 : 0;         // unused; B frag uses hiA: lanes>=16 hold K+8
    const int hiA  = (lane & 16) ? 8  : 0;         // A frag K split / C row split
    const int mA   = lane & 15;                    // A row M

    // ---- preload W_hh^T B-fragments (resident for whole scan) -------------
    v16h bfrag[4];
#pragma unroll
    for (int kc = 0; kc < 4; ++kc) {
        const float* src = W_hh + ncol * H_ + 32 * kc + hiA; // W_hh^T[k][n]=W_hh[n][k]
#pragma unroll
        for (int e = 0; e < 16; ++e) bfrag[kc][e] = (_Float16)src[e + (e & 8)];
    }
    __syncthreads();
    const float bsn = bsum[ncol];

    // ---- the scan: 512 dependent steps ------------------------------------
    for (int t = 0; t < T_; ++t) {
        const int cur = t & 1;

        // one b128 broadcast load fetches all 8 char indices for this step
        union { uint4 q; unsigned short s[8]; } iv;
        iv.q = *(const uint4*)&idxl[t][hiA];       // 16B-aligned (32t + 16*hi)

        // c0 init = W_ih[:, idx] + (b_ih + b_hh)  (the one-hot "GEMM")
        v8f c0;
#pragma unroll
        for (int r = 0; r < 8; ++r)
            c0[r] = (float)WihT[iv.s[r]][ncol] + bsn;
        v8f c1 = {};                               // second accumulator: 2 indep chains

        // A fragments: two ds_load_b128 per K-chunk from padded h buffer
        union { v16h v; uint4 q[2]; } a[4];
        const _Float16* hrow = &hbuf[cur][mA][0];
#pragma unroll
        for (int kc = 0; kc < 4; ++kc) {
            int k0     = 32 * kc + hiA;            // elems 0-7 -> K=k0.., 8-15 -> K=k0+16..
            a[kc].q[0] = *(const uint4*)(hrow + k0);
            a[kc].q[1] = *(const uint4*)(hrow + k0 + 16);
        }

        // h @ W_hh^T : two independent 2-deep WMMA chains
        c0 = __builtin_amdgcn_wmma_f32_16x16x32_f16(false, a[0].v, false, bfrag[0],
                                                    (short)0, c0, false, false);
        c1 = __builtin_amdgcn_wmma_f32_16x16x32_f16(false, a[2].v, false, bfrag[2],
                                                    (short)0, c1, false, false);
        c0 = __builtin_amdgcn_wmma_f32_16x16x32_f16(false, a[1].v, false, bfrag[1],
                                                    (short)0, c0, false, false);
        c1 = __builtin_amdgcn_wmma_f32_16x16x32_f16(false, a[3].v, false, bfrag[3],
                                                    (short)0, c1, false, false);

        // tanh + writeback to the other buffer
#pragma unroll
        for (int r = 0; r < 8; ++r)
            hbuf[cur ^ 1][r + hiA][ncol] = (_Float16)tanh_fast(c0[r] + c1[r]);

        __syncthreads();
    }

    // ---- final FC: out = hT @ W_fc^T + b_fc  (96 = 6 x 16 -> waves 0..5) ---
    if (wave < 6) {
        const int n = wave * 16 + (lane & 15);
        v16h bf[4];
#pragma unroll
        for (int kc = 0; kc < 4; ++kc) {
            const float* src = W_fc + n * H_ + 32 * kc + hiA;
#pragma unroll
            for (int e = 0; e < 16; ++e) bf[kc][e] = (_Float16)src[e + (e & 8)];
        }
        v8f c;
        const float bc = b_fc[n];
#pragma unroll
        for (int r = 0; r < 8; ++r) c[r] = bc;

        const _Float16* hrow = &hbuf[T_ & 1][mA][0]; // final h lives in buffer 0
        union { v16h v; uint4 q[2]; } a;
#pragma unroll
        for (int kc = 0; kc < 4; ++kc) {
            int k0 = 32 * kc + hiA;
            a.q[0] = *(const uint4*)(hrow + k0);
            a.q[1] = *(const uint4*)(hrow + k0 + 16);
            c = __builtin_amdgcn_wmma_f32_16x16x32_f16(false, a.v, false, bf[kc],
                                                       (short)0, c, false, false);
        }
#pragma unroll
        for (int r = 0; r < 8; ++r)
            out[(size_t)(rb + r + hiA) * V_ + n] = c[r];
    }
}

// ---------------------------------------------------------------------------
extern "C" void kernel_launch(void* const* d_in, const int* in_sizes, int n_in,
                              void* d_out, int out_size, void* d_ws, size_t ws_size,
                              hipStream_t stream)
{
    (void)in_sizes; (void)n_in; (void)out_size; (void)ws_size;
    const float* x    = (const float*)d_in[0];
    const float* W_ih = (const float*)d_in[1];
    const float* b_ih = (const float*)d_in[2];
    const float* W_hh = (const float*)d_in[3];
    const float* b_hh = (const float*)d_in[4];
    const float* W_fc = (const float*)d_in[5];
    const float* b_fc = (const float*)d_in[6];
    float* out = (float*)d_out;
    unsigned short* idx = (unsigned short*)d_ws;   // B*T u16 = 512 KB

    // Phase 1: one-hot -> index (one wave per row; exact grid, no tail)
    onehot_argmax_kernel<<<dim3((B_ * T_) / 8), dim3(256), 0, stream>>>(x, idx);
    // Phase 2: WMMA RNN scan + fused FC
    rnn_scan_kernel<<<dim3(B_ / BT), dim3(256), 0, stream>>>(
        idx, W_ih, b_ih, W_hh, b_hh, W_fc, b_fc, out);
}
